// Spatial_Attention_layer_54348516163928
// MI455X (gfx1250) — matrix operand order repaired and checked by reading, not compiled, not run on hardware
//
#include <hip/hip_runtime.h>
#include <hip/hip_bf16.h>

typedef __attribute__((ext_vector_type(2)))  float     v2f;
typedef __attribute__((ext_vector_type(8)))  float     v8f;
typedef __attribute__((ext_vector_type(16))) _Float16  v16h;

#define Bsz 32
#define Nsz 1024
#define Fsz 64
#define Tsz 24

// ---------------------------------------------------------------------------
// Kernel 1: one wave per (b,n).  lane handles f=lane and f=lane+32.
//   lhs[b,n,t]  = sum_f ( sum_t' x[b,n,f,t']*W1[t'] ) * W2[f,t]
//   rhsT[b,t,n] = sum_f x[b,n,f,t]*W3[f]
// ---------------------------------------------------------------------------
__global__ void k_prep(const float* __restrict__ x,
                       const float* __restrict__ W1,
                       const float* __restrict__ W2,
                       const float* __restrict__ W3,
                       float* __restrict__ lhs,
                       float* __restrict__ rhsT) {
    int lane = threadIdx.x & 31;
    int wave = threadIdx.x >> 5;
    int pair = blockIdx.x * 8 + wave;          // pair = b*N + n
    const float* xp = x + (size_t)pair * (Fsz * Tsz);
    const float* p0 = xp + lane * Tsz;
    const float* p1 = xp + (lane + 32) * Tsz;

    float xr0[Tsz], xr1[Tsz];
    float a0 = 0.f, a1 = 0.f;
#pragma unroll
    for (int t = 0; t < Tsz; ++t) {
        xr0[t] = p0[t]; xr1[t] = p1[t];
        float w1 = W1[t];
        a0 += xr0[t] * w1;  a1 += xr1[t] * w1;
    }
    float w3a = W3[lane], w3b = W3[lane + 32];

    float lhsv = 0.f, rhsv = 0.f;
#pragma unroll
    for (int t = 0; t < Tsz; ++t) {
        float vl = a0 * W2[lane * Tsz + t] + a1 * W2[(lane + 32) * Tsz + t];
        float vr = xr0[t] * w3a + xr1[t] * w3b;
#pragma unroll
        for (int m = 16; m >= 1; m >>= 1) {
            vl += __shfl_xor(vl, m, 32);
            vr += __shfl_xor(vr, m, 32);
        }
        if (lane == t) { lhsv = vl; rhsv = vr; }
    }
    if (lane < Tsz) {
        lhs[(size_t)pair * Tsz + lane] = lhsv;
        int b = pair >> 10, n = pair & (Nsz - 1);
        rhsT[((size_t)b * Tsz + lane) * Nsz + n] = rhsv;
    }
}

// ---------------------------------------------------------------------------
// Kernel 1b: pack B = V_s^T (B[k][i] = V_s[i][k]) into f16 WMMA B-fragment
// order for 16x16x32: lanes 0-15 hold K=0..7,16..23 (col=lane), lanes 16-31 hold
// K=8..15,24..31 (col=lane-16).  Bpack[kb][it][lane][16 halves].
// ---------------------------------------------------------------------------
__global__ void k_packB(const float* __restrict__ Vs, _Float16* __restrict__ Bpack) {
    int t = blockIdx.x * 256 + threadIdx.x;       // 65536 threads
    int lane = t & 31;
    int it   = (t >> 5) & 63;
    int kb   = t >> 11;                            // 0..31
    int i    = it * 16 + (lane & 15);
    int kbase = kb * 32 + (lane >> 4) * 8;
    _Float16* dst = Bpack + (((size_t)kb * 64 + it) * 32 + lane) * 16;
#pragma unroll
    for (int e = 0; e < 16; ++e)
        dst[e] = (_Float16)Vs[(size_t)i * Nsz + kbase + e + (e >= 8 ? 8 : 0)];
}

// ---------------------------------------------------------------------------
// Kernel 2: product + sigmoid, f32 WMMA 16x16x4.
// Block = 1 wave; handles (b, nt[16 rows], kb -> m-cols [kb*32, kb*32+32)).
// D tile is repacked through LDS into the A-fragment (16x32 f16) layout for
// the big GEMM and stored to sighPack[b][nt][kb][lane][16].
// ---------------------------------------------------------------------------
__global__ void k_prod_sig(const float* __restrict__ lhs,
                           const float* __restrict__ rhsT,
                           const float* __restrict__ bs,
                           _Float16* __restrict__ sighPack) {
    __shared__ __attribute__((aligned(16))) _Float16 smem[16 * 32];
    int kb = blockIdx.x;          // 0..31  -> m block
    int nt = blockIdx.y;          // 0..63
    int b  = blockIdx.z;          // 0..31
    int lane = threadIdx.x;
    int hb = lane >> 4;           // half-wave id
    int l15 = lane & 15;
    int m0 = kb * 32;

    const float* lb = lhs + ((size_t)b * Nsz + nt * 16) * Tsz;
    const float* rp = rhsT + (size_t)b * Tsz * Nsz;

    v8f acc0 = {}, acc1 = {};
#pragma unroll
    for (int ks = 0; ks < 6; ++ks) {
        int k0 = ks * 4;
        int kk = k0 + 2 * hb;
        v2f a, b0, b1;
        a.x = lb[l15 * Tsz + kk + 0];          // A 16x4: lane<16 K=0,1 ; lane>=16 K=2,3
        a.y = lb[l15 * Tsz + kk + 1];
        b0.x = rp[(size_t)(kk + 0) * Nsz + m0 + l15];
        b0.y = rp[(size_t)(kk + 1) * Nsz + m0 + l15];
        b1.x = rp[(size_t)(kk + 0) * Nsz + m0 + 16 + l15];
        b1.y = rp[(size_t)(kk + 1) * Nsz + m0 + 16 + l15];
        acc0 = __builtin_amdgcn_wmma_f32_16x16x4_f32(false, a, false, b0,
                                                     (short)0, acc0, false, false);
        acc1 = __builtin_amdgcn_wmma_f32_16x16x4_f32(false, a, false, b1,
                                                     (short)0, acc1, false, false);
    }
    // epilogue: + b_s, sigmoid, scatter into LDS 16x32 f16 tile
#pragma unroll
    for (int r = 0; r < 8; ++r) {
        int row = r + hb * 8;
        {
            int col = l15;
            float v = acc0[r] + bs[(size_t)(nt * 16 + row) * Nsz + m0 + col];
            smem[row * 32 + col] = (_Float16)(1.0f / (1.0f + __expf(-v)));
        }
        {
            int col = 16 + l15;
            float v = acc1[r] + bs[(size_t)(nt * 16 + row) * Nsz + m0 + col];
            smem[row * 32 + col] = (_Float16)(1.0f / (1.0f + __expf(-v)));
        }
    }
    __syncthreads();
    // gather this lane's A-fragment (K 0..7 & 16..23 for lanes<16; +8 else)
    int row = l15, c0 = hb * 8;
    const uint4* s0 = (const uint4*)&smem[row * 32 + c0];
    const uint4* s1 = (const uint4*)&smem[row * 32 + c0 + 16];
    uint4* dst = (uint4*)(sighPack + ((((size_t)b * 64 + nt) * 32 + kb) * 32 + lane) * 16);
    dst[0] = *s0;
    dst[1] = *s1;
}

// ---------------------------------------------------------------------------
// Kernel 3: S = sig @ V_s^T, f16 WMMA 16x16x32, f32 accumulate.
// Block = 4 waves; block tile 64 (rows) x 128 (cols).  Per k-block the 8
// B-fragments (8 KB) are staged in LDS (ping-pong, one barrier/iter) and
// shared by all 4 waves; each wave holds 8 accumulators (16x128) and does
// 8 v_wmma per k-block (256 per block iteration space).
// S written to d_out (f32, row-major (B,N,N)).
// ---------------------------------------------------------------------------
__global__ void k_gemm(const _Float16* __restrict__ sighPack,
                       const _Float16* __restrict__ Bpack,
                       float* __restrict__ S) {
    __shared__ __attribute__((aligned(16))) _Float16 Bsm[2][8 * 32 * 16];  // 2 x 8 KB
    int itG = blockIdx.x;         // 0..7   -> i cols [itG*128, +128)
    int ntG = blockIdx.y;         // 0..15  -> rows  [ntG*64,  +64) within b
    int b   = blockIdx.z;         // 0..31
    int tid  = threadIdx.x;
    int wave = tid >> 5;
    int lane = tid & 31;
    int nt  = ntG * 4 + wave;     // this wave's 16-row tile id (0..63)
    int it0 = itG * 8;            // first of 8 i-tiles for this block

    const v16h* Ap = (const v16h*)sighPack + (((size_t)b * 64 + nt) * 32) * 32 + lane;

    v8f acc[8] = {{}, {}, {}, {}, {}, {}, {}, {}};
    for (int kb = 0; kb < 32; ++kb) {
        int p = kb & 1;
        // cooperative stage of 8 B-fragments (256 fragment-lanes, 2 per thread)
#pragma unroll
        for (int s = 0; s < 2; ++s) {
            int slot = tid + s * 128;
            int j = slot >> 5, l = slot & 31;
            const uint4* sp = (const uint4*)(Bpack + (((size_t)kb * 64 + it0 + j) * 32 + l) * 16);
            uint4* dp = (uint4*)(Bsm[p] + ((size_t)j * 32 + l) * 16);
            dp[0] = sp[0];
            dp[1] = sp[1];
        }
        v16h a = Ap[(size_t)kb * 32];
        if (kb + 1 < 32) __builtin_prefetch(&Ap[(size_t)(kb + 1) * 32], 0, 1);
        __syncthreads();
#pragma unroll
        for (int j = 0; j < 8; ++j) {
            v16h bb = *(const v16h*)(Bsm[p] + ((size_t)j * 32 + lane) * 16);
            acc[j] = __builtin_amdgcn_wmma_f32_16x16x32_f16(false, a, false, bb,
                                                            (short)0, acc[j], false, false);
        }
    }
    // D layout: VGPR r -> (M=r, col=lane) for lanes<16 ; (M=r+8, col=lane-16)
    int hb = lane >> 4, l15 = lane & 15;
#pragma unroll
    for (int j = 0; j < 8; ++j) {
        int i = (it0 + j) * 16 + l15;
#pragma unroll
        for (int r = 0; r < 8; ++r) {
            int n = nt * 16 + r + hb * 8;
            S[((size_t)b * Nsz + n) * Nsz + i] = acc[j][r];
        }
    }
}

// ---------------------------------------------------------------------------
// Reductions / normalization
// ---------------------------------------------------------------------------
__global__ void k_colmax_part(const float* __restrict__ S, float* __restrict__ part) {
    int i = blockIdx.x * 256 + threadIdx.x;   // 0..1023
    int ch = blockIdx.y;                      // 0..15
    int b  = blockIdx.z;
    float m = -3.4e38f;
    const float* p = S + ((size_t)b * Nsz + ch * 64) * Nsz + i;
    for (int n = 0; n < 64; ++n) m = fmaxf(m, p[(size_t)n * Nsz]);
    part[((size_t)b * 16 + ch) * Nsz + i] = m;
}

__global__ void k_colmax_fin(const float* __restrict__ part, float* __restrict__ maxbuf) {
    int i = blockIdx.x * 256 + threadIdx.x;
    int b = blockIdx.y;
    float m = -3.4e38f;
    for (int c = 0; c < 16; ++c) m = fmaxf(m, part[((size_t)b * 16 + c) * Nsz + i]);
    maxbuf[(size_t)b * Nsz + i] = m;
}

__global__ void k_denom_part(const float* __restrict__ S, const float* __restrict__ maxbuf,
                             float* __restrict__ dpart) {
    int i = blockIdx.x * 256 + threadIdx.x;
    int ch = blockIdx.y;
    float mx = maxbuf[i];                     // batch 0 only (reference quirk)
    float s = 0.f;
    const float* p = S + (size_t)(ch * 64) * Nsz + i;
    for (int n = 0; n < 64; ++n) s += __expf(p[(size_t)n * Nsz] - mx);
    dpart[(size_t)ch * Nsz + i] = s;
}

__global__ void k_denom_fin(const float* __restrict__ dpart, float* __restrict__ denom) {
    int i = blockIdx.x * 256 + threadIdx.x;
    float s = 0.f;
    for (int c = 0; c < 16; ++c) s += dpart[(size_t)c * Nsz + i];
    denom[i] = s;
}

__global__ void k_final(float* __restrict__ S, const float* __restrict__ maxbuf,
                        const float* __restrict__ denom) {
    size_t idx = (size_t)blockIdx.x * 256 + threadIdx.x;   // over B*N*N
    int i = (int)(idx & (Nsz - 1));
    int b = (int)(idx >> 20);
    float mx = maxbuf[(size_t)b * Nsz + i];
    float rd = 1.0f / denom[i];
    S[idx] = __expf(S[idx] - mx) * rd;
}

// ---------------------------------------------------------------------------
extern "C" void kernel_launch(void* const* d_in, const int* in_sizes, int n_in,
                              void* d_out, int out_size, void* d_ws, size_t ws_size,
                              hipStream_t stream) {
    const float* x   = (const float*)d_in[0];
    const float* W1  = (const float*)d_in[1];
    const float* W2  = (const float*)d_in[2];
    const float* W3  = (const float*)d_in[3];
    const float* bs  = (const float*)d_in[4];
    const float* Vs  = (const float*)d_in[5];
    float* out = (float*)d_out;

    char* ws = (char*)d_ws;
    size_t off = 0;
    float*    lhs      = (float*)(ws + off);    off += (size_t)Bsz * Nsz * Tsz * 4;   // 3 MB
    float*    rhsT     = (float*)(ws + off);    off += (size_t)Bsz * Tsz * Nsz * 4;   // 3 MB
    _Float16* Bpack    = (_Float16*)(ws + off); off += (size_t)Nsz * Nsz * 2;         // 2 MB
    _Float16* sighPack = (_Float16*)(ws + off); off += (size_t)Bsz * Nsz * Nsz * 2;   // 64 MB
    float*    maxPart  = (float*)(ws + off);    off += (size_t)Bsz * 16 * Nsz * 4;    // 2 MB
    float*    maxbuf   = (float*)(ws + off);    off += (size_t)Bsz * Nsz * 4;
    float*    dpart    = (float*)(ws + off);    off += (size_t)16 * Nsz * 4;
    float*    denom    = (float*)(ws + off);    off += (size_t)Nsz * 4;

    // 1) lhs / rhsT from x   (memory-bound sweep of x)
    k_prep<<<dim3(Bsz * Nsz / 8), dim3(256), 0, stream>>>(x, W1, W2, W3, lhs, rhsT);
    // 1b) pack V_s^T into f16 B-fragments
    k_packB<<<dim3(256), dim3(256), 0, stream>>>(Vs, Bpack);
    // 2) product (f32 WMMA 16x16x4) + b_s + sigmoid -> f16 A-fragments
    k_prod_sig<<<dim3(32, 64, Bsz), dim3(32), 0, stream>>>(lhs, rhsT, bs, sighPack);
    // 3) S = sig @ V_s^T  (f16 WMMA 16x16x32, f32 acc, LDS-tiled) -> d_out
    k_gemm<<<dim3(8, 16, Bsz), dim3(128), 0, stream>>>(sighPack, Bpack, out);
    // 4) softmax-with-quirk: per-(b,i) max over n; batch-0 denominator
    k_colmax_part<<<dim3(4, 16, Bsz), dim3(256), 0, stream>>>(out, maxPart);
    k_colmax_fin<<<dim3(4, Bsz), dim3(256), 0, stream>>>(maxPart, maxbuf);
    k_denom_part<<<dim3(4, 16), dim3(256), 0, stream>>>(out, maxbuf, dpart);
    k_denom_fin<<<dim3(4), dim3(256), 0, stream>>>(dpart, denom);
    k_final<<<dim3((unsigned)((size_t)Bsz * Nsz * Nsz / 256)), dim3(256), 0, stream>>>(out, maxbuf, denom);
}
